// ResidualBlockNoBNDynamic_87806311399737
// MI455X (gfx1250) — compile-verified, matrix operand order checked by reading, not run on hardware
//
#include <hip/hip_runtime.h>
#include <hip/hip_bf16.h>
#include <stdint.h>

typedef __attribute__((ext_vector_type(16))) _Float16 v16h;
typedef __attribute__((ext_vector_type(8)))  float    v8f;

#define Bn 16
#define Cc 64
#define Hh 256
#define Ww 256
#define Kk 10
#define KF 576            // Cc * 9  (GEMM K)
#define WSTD 292          // LDS weight row stride in DWORDs (1168 B, 16B-aligned rows)
#define XPC 130           // x-tile cols: 128 + 2 halo
#define XP  66            // x-tile channel dim padded 64->66 (33-dword stride, conflict-free)

// ---------------------------------------------------------------------------
// Kernel 1: per-sample expert-weight aggregation, f32 -> f16, GEMM layout.
// aggW[b][o][kf], kf = (kh*3+kw)*64 + i  (K-step blocks of 32 are i-contiguous)
// ---------------------------------------------------------------------------
__global__ void agg_weights_kernel(const float* __restrict__ attn,
                                   const float* __restrict__ W1,
                                   const float* __restrict__ W2,
                                   _Float16* __restrict__ aggW1,
                                   _Float16* __restrict__ aggW2) {
    int t = blockIdx.x * blockDim.x + threadIdx.x;
    const int per_layer = Bn * Cc * KF;
    if (t >= 2 * per_layer) return;
    const float* Wsrc = (t < per_layer) ? W1 : W2;
    _Float16*    dst  = (t < per_layer) ? aggW1 : aggW2;
    int r  = (t < per_layer) ? t : t - per_layer;
    int b  = r / (Cc * KF);
    int q  = r % (Cc * KF);
    int o  = q / KF;
    int kf = q % KF;
    int g  = kf >> 6;      // kh*3+kw
    int i  = kf & 63;
    float acc = 0.f;
#pragma unroll
    for (int k = 0; k < Kk; ++k)
        acc += attn[b * Kk + k] * Wsrc[((size_t)((k * Cc + o) * Cc + i)) * 9 + g];
    dst[r] = (_Float16)acc;
}

__global__ void agg_bias_kernel(const float* __restrict__ attn,
                                const float* __restrict__ b1,
                                const float* __restrict__ b2,
                                float* __restrict__ aggb1,
                                float* __restrict__ aggb2) {
    int t = blockIdx.x * blockDim.x + threadIdx.x;
    if (t >= 2 * Bn * Cc) return;
    const float* src = (t < Bn * Cc) ? b1 : b2;
    float*       dst = (t < Bn * Cc) ? aggb1 : aggb2;
    int r = t % (Bn * Cc);
    int b = r / Cc, o = r % Cc;
    float a = 0.f;
#pragma unroll
    for (int k = 0; k < Kk; ++k) a += attn[b * Kk + k] * src[k * Cc + o];
    dst[r] = a;
}

// ---------------------------------------------------------------------------
// Kernel 2: dynamic 3x3 conv as implicit GEMM on v_wmma_f32_16x16x32_f16.
// One block = (b, h, 128-wide w strip): D tile 64(out-ch) x 128(px), K=576.
// 8 wave32s: each owns one M tile x four N tiles (A reused 4x per k-step).
// Weights staged global->LDS via GLOBAL_LOAD_ASYNC_TO_LDS_B128 (ASYNCcnt),
// overlapping the VGPR-path halo stage (f32->f16 convert + w<->chan transpose).
// ---------------------------------------------------------------------------
template <bool IN_F16, bool RELU, bool RESIDUAL, bool OUT_F16>
__global__ __launch_bounds__(256)
void dynconv_wmma_kernel(const void* __restrict__ src_v,
                         const float* __restrict__ resid,
                         const _Float16* __restrict__ aggW,
                         const float* __restrict__ aggb,
                         void* __restrict__ dst_v) {
    __shared__ _Float16 wlds[Cc * WSTD * 2];   // 74,752 B (row stride 1168 B)
    __shared__ _Float16 xlds[3 * XPC * XP];    // 51,480 B  [row][col][chan]

    const int tid  = threadIdx.x;
    const int lane = tid & 31;
    const int wave = tid >> 5;

    const int blk = blockIdx.x;
    const int b   = blk >> 9;           // 256 rows * 2 strips per sample
    const int rm  = blk & 511;
    const int h   = rm >> 1;
    const int w0  = (rm & 1) << 7;

    // Warm caches with the residual tile read in the epilogue (64ch x 128px f32).
    if (RESIDUAL) {
        size_t pb = (((size_t)(b * Cc + (tid & 63)) * Hh) + h) * Ww + w0 + ((tid >> 6) << 4);
        __builtin_prefetch((const char*)resid + pb * 4, 0, 1);        // seg 0..3
        __builtin_prefetch((const char*)resid + (pb + 64) * 4, 0, 1); // seg 4..7
    }

    // ---- stage per-sample f16 weights: async DMA global -> LDS (no VGPR data) ----
    {
        const char* gsrc = (const char*)(aggW + (size_t)b * Cc * KF);
        const unsigned wbase = (unsigned)(uintptr_t)(void*)wlds; // flat[31:0] = LDS offset
        // 64 rows x 72 16B-chunks; dest row stride 1168 B (16B aligned), src contiguous.
        for (int it = 0; it < (Cc * 72) / 256; ++it) {
            int idx = tid + it * 256;
            int o = idx / 72;
            int c = idx - o * 72;
            unsigned long long ga = (unsigned long long)(uintptr_t)(gsrc + idx * 16);
            unsigned           la = wbase + (unsigned)(o * (WSTD * 4) + c * 16);
            asm volatile("global_load_async_to_lds_b128 %0, %1, off"
                         :: "v"(la), "v"(ga) : "memory");
        }
    }

    // ---- stage 3-row input halo, transposed to [row][col][chan] f16 ----
    for (int t = tid; t < 3 * Cc * XPC; t += 256) {
        int col = t % XPC;           // col fastest -> coalesced global reads
        int q   = t / XPC;
        int i   = q & 63;
        int r   = q >> 6;
        int hs  = h + r - 1;
        int ws_ = w0 + col - 1;
        float v = 0.f;
        if ((unsigned)hs < Hh && (unsigned)ws_ < Ww) {
            size_t off = (((size_t)(b * Cc + i) * Hh) + hs) * Ww + ws_;
            v = IN_F16 ? (float)((const _Float16*)src_v)[off]
                       : ((const float*)src_v)[off];
        }
        xlds[(r * XPC + col) * XP + i] = (_Float16)v;
    }
    asm volatile("s_wait_asynccnt 0x0" ::: "memory");  // async DMA done before barrier
    __syncthreads();

    // ---- per-wave WMMA main loop: 1 A-frag x 4 B-frags per k-step ----
    const int mt = wave & 3;              // M tile (out-channel group)
    const int nh = wave >> 2;             // which half of the 8 N tiles
    const int lr = lane & 15;
    const int hi = lane >> 4;             // lane-half selects K sub-block (ISA 7.12.2)

    v8f acc[4] = {v8f{}, v8f{}, v8f{}, v8f{}};

    const unsigned* wq   = (const unsigned*)wlds;
    const int       wrow = (mt * 16 + lr) * WSTD;

#pragma unroll
    for (int ks = 0; ks < 18; ++ks) {     // K = 18 * 32 = 576
        const int g  = ks >> 1;           // kh*3+kw
        const int kh = g / 3;
        const int kw = g - kh * 3;
        const int ih = (ks & 1) << 5;     // input-channel half

        union F { unsigned u[8]; v16h h; } A, B[4];
#pragma unroll
        for (int j = 0; j < 8; ++j) {
            // 16-bit A 16x32 striping: VGPR j holds K pair {2j | 16+2(j-4)} + hi*8
            int kl = ((j < 4) ? (2 * j) : (16 + 2 * (j - 4))) + (hi << 3);
            A.u[j] = wq[wrow + (((ks << 5) + kl) >> 1)];

            int i  = ih + kl;                        // even -> dword-aligned pair
            int c0 = (nh << 6) + lr + kw;            // halo col of N tile 0
#pragma unroll
            for (int nt = 0; nt < 4; ++nt)
                B[nt].u[j] = *(const unsigned*)&xlds[(kh * XPC + c0 + nt * 16) * XP + i];
        }
#pragma unroll
        for (int nt = 0; nt < 4; ++nt)
            acc[nt] = __builtin_amdgcn_wmma_f32_16x16x32_f16(
                false, A.h, false, B[nt].h, (short)0, acc[nt], false, false);
    }

    // ---- epilogue: bias (+ReLU) (+residual), coalesced stores ----
#pragma unroll
    for (int vv = 0; vv < 8; ++vv) {
        int   o    = mt * 16 + vv + (hi << 3);       // D layout: M = vgpr + hi*8
        float bias = aggb[b * Cc + o];
        size_t base = (((size_t)(b * Cc + o) * Hh) + h) * Ww + w0;
#pragma unroll
        for (int nt = 0; nt < 4; ++nt) {
            int   n = (nh << 6) + nt * 16 + lr;
            float r = acc[nt][vv] + bias;
            if (RELU)     r = fmaxf(r, 0.f);
            if (RESIDUAL) r += resid[base + n];
            if (OUT_F16)  ((_Float16*)dst_v)[base + n] = (_Float16)r;
            else          ((float*)dst_v)[base + n] = r;
        }
    }
}

// ---------------------------------------------------------------------------
extern "C" void kernel_launch(void* const* d_in, const int* in_sizes, int n_in,
                              void* d_out, int out_size, void* d_ws, size_t ws_size,
                              hipStream_t stream) {
    const float* x    = (const float*)d_in[0];
    const float* attn = (const float*)d_in[1];
    // d_in[2]=gamma, d_in[3]=beta: pass-through in the reference, unused here.
    const float* W1 = (const float*)d_in[4];
    const float* b1 = (const float*)d_in[5];
    const float* W2 = (const float*)d_in[6];
    const float* b2 = (const float*)d_in[7];
    float* out = (float*)d_out;

    // Workspace layout: aggW1 | aggW2 | aggb1 | aggb2 | mid (f16 activations)
    char* ws = (char*)d_ws;
    _Float16* aggW1 = (_Float16*)ws;
    _Float16* aggW2 = aggW1 + (size_t)Bn * Cc * KF;
    float*    aggb1 = (float*)(ws + 2 * (size_t)Bn * Cc * KF * sizeof(_Float16));
    float*    aggb2 = aggb1 + Bn * Cc;
    _Float16* mid   = (_Float16*)(aggb2 + Bn * Cc);   // 134 MB, written before read

    const int aggN = 2 * Bn * Cc * KF;
    agg_weights_kernel<<<(aggN + 255) / 256, 256, 0, stream>>>(attn, W1, W2, aggW1, aggW2);
    agg_bias_kernel<<<(2 * Bn * Cc + 255) / 256, 256, 0, stream>>>(attn, b1, b2, aggb1, aggb2);

    dim3 grid(Bn * Hh * (Ww / 128));
    // layer 1: x(f32) -> mid(f16), ReLU
    dynconv_wmma_kernel<false, true, false, true>
        <<<grid, 256, 0, stream>>>(x, nullptr, aggW1, aggb1, mid);
    // layer 2: mid(f16) -> out(f32), + residual x
    dynconv_wmma_kernel<true, false, true, false>
        <<<grid, 256, 0, stream>>>(mid, x, aggW2, aggb2, out);
}
